// Model_72387378807327
// MI455X (gfx1250) — matrix operand order repaired and checked by reading, not compiled, not run on hardware
//
#include <hip/hip_runtime.h>
#include <math.h>

#define BATCH 32
#define NPTS  8192
#define N_C   256
#define N_P   32

typedef __attribute__((ext_vector_type(16))) _Float16 v16h;
typedef __attribute__((ext_vector_type(8)))  float    v8f;

union BFrag { unsigned int u[8]; v16h h; };

// XOR swizzle (dword granularity) to break 64-bank row-stride conflicts in the
// LDS activation buffer. Must be used identically by writer and reader.
#define SW(row, kp) ((kp) ^ (((row) & 15) << 2))

__device__ inline unsigned int pack2(float a, float b) {
    union { _Float16 h[2]; unsigned int u; } v;
    v.h[0] = (_Float16)a; v.h[1] = (_Float16)b;
    return v.u;
}

__device__ inline v8f wmma_f16(v16h a, v16h b, v8f c) {
    // v_wmma_f32_16x16x32_f16  (8-arg form: neg_a, A, neg_b, B, c_mod, C, reuse_a, reuse_b)
    return __builtin_amdgcn_wmma_f32_16x16x32_f16(false, a, false, b, (short)0, c, false, false);
}

// A fragment (16x32 f16) from LDS activation buffer.
// Layout per ISA: lanes 0-15 -> M=lane, regs 0-3: K=0..7, regs 4-7: K=16..23;
// lanes 16-31 -> M=lane-16, regs 0-3: K=8..15, regs 4-7: K=24..31.
__device__ inline v16h load_a(const unsigned int* actU, int KAd, int rowtile, int kc, int lane) {
    int row = rowtile * 16 + (lane & 15);
    int h   = lane >> 4;
    int base = kc * 16 + h * 4;  // dword index of first K pair
    BFrag f;
#pragma unroll
    for (int j = 0; j < 4; ++j) {
        f.u[j]     = actU[row * KAd + SW(row, base + j)];
        f.u[4 + j] = actU[row * KAd + SW(row, base + 8 + j)];
    }
    return f.h;
}

// B fragment (32x16 f16) from pre-packed weights: 8 coalesced dwords per lane.
__device__ inline v16h load_b(const unsigned int* packed, int NT, int kt, int nt, int lane) {
    const unsigned int* p = packed + (((size_t)(kt * NT + nt) * 32) + lane) * 8;
    BFrag f;
#pragma unroll
    for (int j = 0; j < 8; ++j) f.u[j] = p[j];
    return f.h;
}

// ---------------------------------------------------------------------------
// Init: zero the global scale accumulator (atomicMax target).
// ---------------------------------------------------------------------------
__global__ void init_kernel(unsigned int* scale_bits) {
    if (threadIdx.x == 0 && blockIdx.x == 0) scale_bits[0] = 0u;
}

// ---------------------------------------------------------------------------
// Pack a row-major f32 weight [K,N] into WMMA-B fragment order (f16 pairs).
// B layout: lane<16 -> N=lane, V0-3: K=0..7, V4-7: K=16..23; lane>=16 -> V0-3: K=8..15, V4-7: K=24..31.
// ---------------------------------------------------------------------------
__global__ void __launch_bounds__(256) pack_b_kernel(const float* W, int K, int N, unsigned int* out) {
    int NT = N >> 4;
    int total = (K >> 5) * NT * 32 * 8;
    for (int idx = blockIdx.x * blockDim.x + threadIdx.x; idx < total; idx += gridDim.x * blockDim.x) {
        int j    = idx & 7;
        int lane = (idx >> 3) & 31;
        int tile = idx >> 8;
        int nt = tile % NT;
        int kt = tile / NT;
        int n = nt * 16 + (lane & 15);
        int k = kt * 32 + ((lane >> 4) << 3) + ((j & 4) << 2) + 2 * (j & 3);
        out[idx] = pack2(W[(size_t)k * N + n], W[(size_t)(k + 1) * N + n]);
    }
}

// ---------------------------------------------------------------------------
// Farthest point sampling: one workgroup per batch, mind[] lives in LDS.
// Tie-break = lowest index (matches jnp.argmax).
// ---------------------------------------------------------------------------
__global__ void __launch_bounds__(256) fps_kernel(const float* data, const int* perm, float* centers) {
    __shared__ float mind[NPTS];
    __shared__ float redV[256];
    __shared__ int   redI[256];
    __shared__ float cx[N_C], cy[N_C], cz[N_C];
    __shared__ float cur[3];

    int b = blockIdx.x, t = threadIdx.x;
    const float* px = data + (size_t)b * 3 * NPTS;
    const float* py = px + NPTS;
    const float* pz = py + NPTS;

    if (t == 0) {
        cur[0] = px[0]; cur[1] = py[0]; cur[2] = pz[0];
        cx[0] = cur[0]; cy[0] = cur[1]; cz[0] = cur[2];
    }
    __syncthreads();
    {
        float c0 = cur[0], c1 = cur[1], c2 = cur[2];
        for (int n = t; n < NPTS; n += 256) {
            float dx = px[n] - c0, dy = py[n] - c1, dz = pz[n] - c2;
            mind[n] = dx * dx + dy * dy + dz * dz;
        }
    }
    __syncthreads();

    for (int i = 1; i < N_C; ++i) {
        float bv = -1.0f; int bi = 0;
        for (int n = t; n < NPTS; n += 256) {
            float v = mind[n];
            if (v > bv || (v == bv && n < bi)) { bv = v; bi = n; }
        }
        redV[t] = bv; redI[t] = bi;
        __syncthreads();
        for (int s = 128; s > 0; s >>= 1) {
            if (t < s) {
                float v = redV[t + s]; int ii = redI[t + s];
                if (v > redV[t] || (v == redV[t] && ii < redI[t])) { redV[t] = v; redI[t] = ii; }
            }
            __syncthreads();
        }
        if (t == 0) {
            int far = redI[0];
            float fx = px[far], fy = py[far], fz = pz[far];
            cur[0] = fx; cur[1] = fy; cur[2] = fz;
            cx[i] = fx; cy[i] = fy; cz[i] = fz;
        }
        __syncthreads();
        float c0 = cur[0], c1 = cur[1], c2 = cur[2];
        for (int n = t; n < NPTS; n += 256) {
            float dx = px[n] - c0, dy = py[n] - c1, dz = pz[n] - c2;
            float d = dx * dx + dy * dy + dz * dz;
            mind[n] = fminf(mind[n], d);
        }
        __syncthreads();
    }

    if (t < N_C) {
        int src = perm[t];
        centers[(size_t)(b * N_C + t) * 3 + 0] = cx[src];
        centers[(size_t)(b * N_C + t) * 3 + 1] = cy[src];
        centers[(size_t)(b * N_C + t) * 3 + 2] = cz[src];
    }
}

// ---------------------------------------------------------------------------
// KNN + local coords + group mean + global max-norm^2 (scale).
// One workgroup per (b,c) row; dist[8192] in LDS; 32-pass argmin selection.
// ---------------------------------------------------------------------------
__global__ void __launch_bounds__(256) knn_kernel(const float* data, const float* centers,
                                                  float* localbuf, float* meanbuf,
                                                  unsigned int* scale_bits) {
    __shared__ float dist[NPTS];
    __shared__ float redV[256];
    __shared__ int   redI[256];
    __shared__ int   knn[N_P];
    __shared__ float nn[N_P * 3];
    __shared__ float mean_s[3];

    int row = blockIdx.x;         // b * N_C + c
    int b = row >> 8;
    int t = threadIdx.x;
    const float* px = data + (size_t)b * 3 * NPTS;
    const float* py = px + NPTS;
    const float* pz = py + NPTS;
    float c0 = centers[(size_t)row * 3 + 0];
    float c1 = centers[(size_t)row * 3 + 1];
    float c2 = centers[(size_t)row * 3 + 2];

    for (int n = t; n < NPTS; n += 256) {
        float dx = px[n] - c0, dy = py[n] - c1, dz = pz[n] - c2;
        dist[n] = dx * dx + dy * dy + dz * dz;
    }
    __syncthreads();

    for (int i = 0; i < N_P; ++i) {
        float bv = 3.4e38f; int bi = NPTS;
        for (int n = t; n < NPTS; n += 256) {
            float v = dist[n];
            if (v < bv || (v == bv && n < bi)) { bv = v; bi = n; }
        }
        redV[t] = bv; redI[t] = bi;
        __syncthreads();
        for (int s = 128; s > 0; s >>= 1) {
            if (t < s) {
                float v = redV[t + s]; int ii = redI[t + s];
                if (v < redV[t] || (v == redV[t] && ii < redI[t])) { redV[t] = v; redI[t] = ii; }
            }
            __syncthreads();
        }
        if (t == 0) {
            knn[i] = redI[0];
            dist[redI[0]] = 3.4e38f;
        }
        __syncthreads();
    }

    if (t < N_P) {
        int n = knn[t];
        nn[t * 3 + 0] = px[n]; nn[t * 3 + 1] = py[n]; nn[t * 3 + 2] = pz[n];
    }
    __syncthreads();
    if (t < 3) {
        float s = 0.0f;
        for (int i = 0; i < N_P; ++i) s += nn[i * 3 + t];
        float m = s * (1.0f / N_P);
        mean_s[t] = m;
        meanbuf[(size_t)row * 3 + t] = m;
    }
    __syncthreads();
    if (t < N_P) {
        float lx = nn[t * 3 + 0] - mean_s[0];
        float ly = nn[t * 3 + 1] - mean_s[1];
        float lz = nn[t * 3 + 2] - mean_s[2];
        size_t o = ((size_t)row * N_P + t) * 3;
        localbuf[o + 0] = lx; localbuf[o + 1] = ly; localbuf[o + 2] = lz;
        float nsq = lx * lx + ly * ly + lz * lz;
        atomicMax(scale_bits, __float_as_uint(nsq));
    }
}

// ---------------------------------------------------------------------------
// FoldingNet decoder: one workgroup (8 waves) per patch.
// WMMA (f16 in / f32 acc) for the three [32,K]x[K,N] GEMMs; codeword GEMMs
// collapse to per-patch vector-matrix products (rows share the codeword).
// ---------------------------------------------------------------------------
__global__ void __launch_bounds__(256) decoder_kernel(
    const float* grid, const float* We1, const float* be1, const float* be2,
    const float* Wf1a, const float* bf1a, const float* bf1b,
    const float* Wf1c, const float* bf1c,
    const float* Wf2a, const float* bf2a, const float* bf2b,
    const float* Wf2c, const float* bf2c,
    const unsigned int* packWe2, const unsigned int* packWf1b, const unsigned int* packWf2b,
    const float* localbuf, const float* meanbuf, const float* scale_sq,
    float* out)
{
    __shared__ float        xbuf[N_P * 3];
    __shared__ unsigned int actU[N_P * 128];   // 32 rows x up to 256 f16 (128 dwords/row)
    __shared__ float        code[256];
    __shared__ float        basev[256];
    __shared__ float        gbuf[N_P * 128];   // f32 activations for the x3 output layers
    __shared__ float        f3[N_P * 3];
    __shared__ float        colmax[256 * 2];

    int t    = threadIdx.x;
    int lane = t & 31;
    int wave = t >> 5;
    int p    = blockIdx.x;                     // b * N_C + c

    float scl = sqrtf(scale_sq[0]);
    float inv = 1.0f / scl;

    if (t < N_P * 3) xbuf[t] = localbuf[(size_t)p * N_P * 3 + t] * inv;
    for (int i = t; i < 512; i += 256) colmax[i] = -1e30f;
    __syncthreads();

    // ---- L1: h1[32][128] = relu(x @ We1 + be1) -> f16 act buffer (KAd = 64)
    for (int d = t; d < N_P * 64; d += 256) {
        int row = d >> 6;
        int kp  = d & 63;
        int k0 = kp * 2, k1 = k0 + 1;
        float x0 = xbuf[row * 3 + 0], x1 = xbuf[row * 3 + 1], x2 = xbuf[row * 3 + 2];
        float a = fmaf(x0, We1[k0], fmaf(x1, We1[128 + k0], fmaf(x2, We1[256 + k0], be1[k0])));
        float b = fmaf(x0, We1[k1], fmaf(x1, We1[128 + k1], fmaf(x2, We1[256 + k1], be1[k1])));
        actU[row * 64 + SW(row, kp)] = pack2(fmaxf(a, 0.0f), fmaxf(b, 0.0f));
    }
    __syncthreads();

    // ---- L2 (WMMA): h2 = h1 @ We2 ; never stored — fold straight into column max.
    {
#pragma unroll
        for (int q = 0; q < 2; ++q) {
            int nt = wave * 2 + q;
            float pmax = -1e30f;
#pragma unroll
            for (int r = 0; r < 2; ++r) {
                v8f acc = {};
#pragma unroll
                for (int kc = 0; kc < 4; ++kc) {
                    v16h a = load_a(actU, 64, r, kc, lane);
                    v16h b = load_b(packWe2, 16, kc, nt, lane);
                    acc = wmma_f16(a, b, acc);
                }
#pragma unroll
                for (int j = 0; j < 8; ++j) pmax = fmaxf(pmax, acc[j]);
            }
            colmax[(nt * 16 + (lane & 15)) * 2 + (lane >> 4)] = pmax;
        }
    }
    __syncthreads();
    // code[n] = relu(max_rows + be2)   (relu/bias commute with row-max)
    code[t] = fmaxf(fmaxf(colmax[t * 2], colmax[t * 2 + 1]) + be2[t], 0.0f);
    __syncthreads();

    // ---- L3 base: base1[n] = code . Wf1a[0:256, n] + bf1a[n]  (shared by all rows)
    {
        float s = bf1a[t];
        for (int k = 0; k < 256; ++k) s = fmaf(code[k], Wf1a[(size_t)k * 256 + t], s);
        basev[t] = s;
    }
    __syncthreads();
    // f1[32][256] = relu(base1 + grid terms) -> f16 act buffer (KAd = 128)
    for (int d = t; d < N_P * 128; d += 256) {
        int row = d >> 7;
        int kp  = d & 127;
        int n0 = kp * 2, n1 = n0 + 1;
        float g0 = grid[row * 2 + 0], g1 = grid[row * 2 + 1];
        float a = basev[n0] + g0 * Wf1a[256 * 256 + n0] + g1 * Wf1a[257 * 256 + n0];
        float b = basev[n1] + g0 * Wf1a[256 * 256 + n1] + g1 * Wf1a[257 * 256 + n1];
        actU[row * 128 + SW(row, kp)] = pack2(fmaxf(a, 0.0f), fmaxf(b, 0.0f));
    }
    __syncthreads();

    // ---- L4 (WMMA): g1[32][128] = relu(f1 @ Wf1b + bf1b)
    {
        int nt = wave;
#pragma unroll
        for (int r = 0; r < 2; ++r) {
            v8f acc = {};
#pragma unroll
            for (int kc = 0; kc < 8; ++kc) {
                v16h a = load_a(actU, 128, r, kc, lane);
                v16h b = load_b(packWf1b, 8, kc, nt, lane);
                acc = wmma_f16(a, b, acc);
            }
            int col = nt * 16 + (lane & 15);
            float bv = bf1b[col];
#pragma unroll
            for (int j = 0; j < 8; ++j) {
                int row = r * 16 + ((lane >> 4) << 3) + j;
                gbuf[row * 128 + col] = fmaxf(acc[j] + bv, 0.0f);
            }
        }
    }
    __syncthreads();

    // ---- L5: f3[32][3] = g1 @ Wf1c + bf1c (no relu)
    if (t < 96) {
        int row = t / 3, cc = t % 3;
        float s = bf1c[cc];
        for (int k = 0; k < 128; ++k) s = fmaf(gbuf[row * 128 + k], Wf1c[k * 3 + cc], s);
        f3[row * 3 + cc] = s;
    }
    __syncthreads();

    // ---- L6 base: base2[n] = code . Wf2a[0:256, n] + bf2a[n]
    {
        float s = bf2a[t];
        for (int k = 0; k < 256; ++k) s = fmaf(code[k], Wf2a[(size_t)k * 256 + t], s);
        basev[t] = s;
    }
    __syncthreads();
    // f2a[32][256] = relu(base2 + fold terms) -> f16 act buffer
    for (int d = t; d < N_P * 128; d += 256) {
        int row = d >> 7;
        int kp  = d & 127;
        int n0 = kp * 2, n1 = n0 + 1;
        float y0 = f3[row * 3 + 0], y1 = f3[row * 3 + 1], y2 = f3[row * 3 + 2];
        float a = basev[n0] + y0 * Wf2a[256 * 256 + n0] + y1 * Wf2a[257 * 256 + n0] + y2 * Wf2a[258 * 256 + n0];
        float b = basev[n1] + y0 * Wf2a[256 * 256 + n1] + y1 * Wf2a[257 * 256 + n1] + y2 * Wf2a[258 * 256 + n1];
        actU[row * 128 + SW(row, kp)] = pack2(fmaxf(a, 0.0f), fmaxf(b, 0.0f));
    }
    __syncthreads();

    // ---- L7 (WMMA): g2[32][128] = relu(f2a @ Wf2b + bf2b)
    {
        int nt = wave;
#pragma unroll
        for (int r = 0; r < 2; ++r) {
            v8f acc = {};
#pragma unroll
            for (int kc = 0; kc < 8; ++kc) {
                v16h a = load_a(actU, 128, r, kc, lane);
                v16h b = load_b(packWf2b, 8, kc, nt, lane);
                acc = wmma_f16(a, b, acc);
            }
            int col = nt * 16 + (lane & 15);
            float bv = bf2b[col];
#pragma unroll
            for (int j = 0; j < 8; ++j) {
                int row = r * 16 + ((lane >> 4) << 3) + j;
                gbuf[row * 128 + col] = fmaxf(acc[j] + bv, 0.0f);
            }
        }
    }
    __syncthreads();

    // ---- L8: out[32][3] = (g2 @ Wf2c + bf2c) * scale + mean
    if (t < 96) {
        int row = t / 3, cc = t % 3;
        float s = bf2c[cc];
        for (int k = 0; k < 128; ++k) s = fmaf(gbuf[row * 128 + k], Wf2c[k * 3 + cc], s);
        out[(size_t)p * N_P * 3 + row * 3 + cc] = s * scl + meanbuf[(size_t)p * 3 + cc];
    }
}

// ---------------------------------------------------------------------------
extern "C" void kernel_launch(void* const* d_in, const int* in_sizes, int n_in,
                              void* d_out, int out_size, void* d_ws, size_t ws_size,
                              hipStream_t stream) {
    (void)in_sizes; (void)n_in; (void)out_size; (void)ws_size;

    const float* data = (const float*)d_in[0];
    const int*   perm = (const int*)d_in[1];
    const float* grid = (const float*)d_in[2];
    const float* We1  = (const float*)d_in[3];
    const float* be1  = (const float*)d_in[4];
    const float* We2  = (const float*)d_in[5];
    const float* be2  = (const float*)d_in[6];
    const float* Wf1a = (const float*)d_in[7];
    const float* bf1a = (const float*)d_in[8];
    const float* Wf1b = (const float*)d_in[9];
    const float* bf1b = (const float*)d_in[10];
    const float* Wf1c = (const float*)d_in[11];
    const float* bf1c = (const float*)d_in[12];
    const float* Wf2a = (const float*)d_in[13];
    const float* bf2a = (const float*)d_in[14];
    const float* Wf2b = (const float*)d_in[15];
    const float* bf2b = (const float*)d_in[16];
    const float* Wf2c = (const float*)d_in[17];
    const float* bf2c = (const float*)d_in[18];

    // Workspace layout (floats / uints)
    float* ws = (float*)d_ws;
    unsigned int* scale_bits = (unsigned int*)ws;                 // [0]
    float* centers  = ws + 16;                                    // 32*256*3
    float* meanbuf  = centers + BATCH * N_C * 3;                  // 32*256*3
    float* localbuf = meanbuf + BATCH * N_C * 3;                  // 32*256*32*3
    unsigned int* packWe2  = (unsigned int*)(localbuf + (size_t)BATCH * N_C * N_P * 3);
    unsigned int* packWf1b = packWe2  + (128 * 256) / 2;
    unsigned int* packWf2b = packWf1b + (256 * 128) / 2;

    init_kernel<<<1, 1, 0, stream>>>(scale_bits);
    fps_kernel<<<BATCH, 256, 0, stream>>>(data, perm, centers);
    knn_kernel<<<BATCH * N_C, 256, 0, stream>>>(data, centers, localbuf, meanbuf, scale_bits);
    pack_b_kernel<<<64, 256, 0, stream>>>(We2, 128, 256, packWe2);
    pack_b_kernel<<<64, 256, 0, stream>>>(Wf1b, 256, 128, packWf1b);
    pack_b_kernel<<<64, 256, 0, stream>>>(Wf2b, 256, 128, packWf2b);
    decoder_kernel<<<BATCH * N_C, 256, 0, stream>>>(
        grid, We1, be1, be2, Wf1a, bf1a, bf1b, Wf1c, bf1c,
        Wf2a, bf2a, bf2b, Wf2c, bf2c,
        packWe2, packWf1b, packWf2b,
        localbuf, meanbuf, (const float*)scale_bits, (float*)d_out);
}
